// FlashAttention_20109036880140
// MI455X (gfx1250) — matrix operand order repaired and checked by reading, not compiled, not run on hardware
//
#include <hip/hip_runtime.h>

typedef __attribute__((ext_vector_type(16))) __bf16 v16bf;
typedef __attribute__((ext_vector_type(8)))  __bf16 v8bf;
typedef __attribute__((ext_vector_type(8)))  float  v8f;

#define NH 16
#define HD 64
#define DM 1024          // NH * HD
#define BM 64            // query rows per workgroup
#define BN 64            // keys per KV tile
#define NWAVES 4

static __device__ inline unsigned short f2bfbits(float f) {
    unsigned u = __builtin_bit_cast(unsigned, f);
    u += 0x7FFFu + ((u >> 16) & 1u);          // round-to-nearest-even
    return (unsigned short)(u >> 16);
}
static __device__ inline __bf16 f2bf(float f) {
    unsigned short b = f2bfbits(f);
    return __builtin_bit_cast(__bf16, b);
}
static __device__ inline unsigned pk2(float lo, float hi) {
    return (unsigned)f2bfbits(lo) | ((unsigned)f2bfbits(hi) << 16);
}
static __device__ inline v16bf cat16(v8bf lo, v8bf hi) {
    v16bf r;
#pragma unroll
    for (int i = 0; i < 8; ++i) { r[i] = lo[i]; r[i + 8] = hi[i]; }
    return r;
}
// 2 x 8 bf16 at p and p+16 (B-operand, table k order), 16B aligned
static __device__ inline v16bf ld_b16(const __bf16* p) {
    v8bf lo = *(const v8bf*)p;
    v8bf hi = *(const v8bf*)(p + 16);
    return cat16(lo, hi);
}
static __device__ inline v8f wmma_bf(v16bf a, v16bf b, v8f c) {
    return __builtin_amdgcn_wmma_f32_16x16x32_bf16(false, a, false, b, (short)0, c,
                                                   false, false);
}

// ---------------- prep kernel 1: K f32 -> bf16 (same layout) ----------------
__global__ __launch_bounds__(256)
void cvt_k_kernel(const float* __restrict__ k, unsigned short* __restrict__ kb) {
    int g = (blockIdx.x * 256 + threadIdx.x) * 4;
    const float4 x = *(const float4*)(k + g);
    uint2 st; st.x = pk2(x.x, x.y); st.y = pk2(x.z, x.w);
    *(uint2*)(kb + g) = st;
}

// ------- prep kernel 2: V f32 -> bf16, transposed per head: vt[h*64+d][s] ----
__global__ __launch_bounds__(256)
void vtrans_kernel(const float* __restrict__ v, unsigned short* __restrict__ vt, int S) {
    __shared__ unsigned short tile[64][72];      // padded to dodge bank conflicts
    const int s0 = blockIdx.x * 64, d0 = blockIdx.y * 64;
    const int tid = threadIdx.x;
    const int rl = tid >> 2, cq = tid & 3;
    const float* src = v + (size_t)(s0 + rl) * DM + d0 + cq * 16;
#pragma unroll
    for (int j = 0; j < 4; ++j) {
        float4 x = *(const float4*)(src + j * 4);
        tile[rl][cq * 16 + j * 4 + 0] = f2bfbits(x.x);
        tile[rl][cq * 16 + j * 4 + 1] = f2bfbits(x.y);
        tile[rl][cq * 16 + j * 4 + 2] = f2bfbits(x.z);
        tile[rl][cq * 16 + j * 4 + 3] = f2bfbits(x.w);
    }
    __syncthreads();
    unsigned w[8];
#pragma unroll
    for (int i = 0; i < 8; ++i) {
        unsigned lo = tile[cq * 16 + 2 * i + 0][rl];
        unsigned hi = tile[cq * 16 + 2 * i + 1][rl];
        w[i] = lo | (hi << 16);
    }
    unsigned short* dst = vt + (size_t)(d0 + rl) * S + s0 + cq * 16;
    uint4 a; a.x = w[0]; a.y = w[1]; a.z = w[2]; a.w = w[3];
    uint4 b; b.x = w[4]; b.y = w[5]; b.z = w[6]; b.w = w[7];
    *(uint4*)dst       = a;
    *((uint4*)dst + 1) = b;
}

// ---------------- main attention kernel ----------------
// PRE=1: K/V pre-converted bf16 in global; async-to-LDS double-buffered fill.
// PRE=0: self-contained (convert f32->bf16 in-kernel, synchronous fill).
template <bool PRE>
__global__ __launch_bounds__(128, 1)
void fa_fwd_kernel(const float* __restrict__ q,
                   const float* __restrict__ kf, const float* __restrict__ vf,
                   const __bf16* __restrict__ kb, const __bf16* __restrict__ vtb,
                   float* __restrict__ out, int S) {
    __shared__ __attribute__((aligned(16))) __bf16 lds_k[2][BN * HD];   // [key][dim]
    __shared__ __attribute__((aligned(16))) __bf16 lds_v[2][HD * BN];   // [dim][key]
    __shared__ __attribute__((aligned(16))) __bf16 lds_p[NWAVES * 16 * BN];

    const int tid  = threadIdx.x;
    const int wave = tid >> 5;
    const int lane = tid & 31;
    const int ln   = lane & 15;
    const int hs   = lane >> 4;
    const int qb   = blockIdx.x;
    const int h    = blockIdx.y;
    const int qrow = qb * BM + wave * 16 + ln;

    // ---- Q as A-operands (bf16), loaded once ----
    v16bf qa[2];
#pragma unroll
    for (int c = 0; c < 2; ++c) {
        const float* qp = q + (size_t)qrow * DM + h * HD + c * 32 + hs * 8;
#pragma unroll
        for (int i = 0; i < 8; ++i) {
            qa[c][i]     = f2bf(qp[i]);
            qa[c][i + 8] = f2bf(qp[16 + i]);
        }
    }

    v8f O[4];
#pragma unroll
    for (int t = 0; t < 4; ++t) { v8f z = {}; O[t] = z; }
    float mi[8], li[8];
#pragma unroll
    for (int r = 0; r < 8; ++r) { mi[r] = -1.0e30f; li[r] = 0.0f; }

    const float scale = 0.125f;   // 64^-0.5
    const int nIter = S / BN;

    // issue one KV tile (8 KB K + 8 KB Vt) as 16B-per-lane async copies
    auto issue_tile = [&](int buf, int itx) {
        const int kv0 = itx * BN;
#pragma unroll
        for (int i = 0; i < 4; ++i) {
            int f = (i * 128 + tid) * 8;          // 8 bf16 = 16 bytes per lane
            int r = f >> 6, c = f & 63;
            const __bf16* gk = kb  + (size_t)(kv0 + r) * DM + h * HD + c;
            const __bf16* gv = vtb + (size_t)(h * HD + r) * S + kv0 + c;
            unsigned lk = (unsigned)(size_t)&lds_k[buf][f];
            unsigned lv = (unsigned)(size_t)&lds_v[buf][f];
            asm volatile("global_load_async_to_lds_b128 %0, %1, off"
                         :: "v"(lk), "v"(gk) : "memory");
            asm volatile("global_load_async_to_lds_b128 %0, %1, off"
                         :: "v"(lv), "v"(gv) : "memory");
        }
    };

    if constexpr (PRE) issue_tile(0, 0);

    for (int it = 0; it < nIter; ++it) {
        const int cur = it & 1;
        const int kv0 = it * BN;

        if constexpr (PRE) {
            if (it + 1 < nIter) {
                issue_tile(cur ^ 1, it + 1);                       // prefetch next tile
                asm volatile("s_wait_asynccnt 8" ::: "memory");    // current tile done
            } else {
                asm volatile("s_wait_asynccnt 0" ::: "memory");
            }
        } else {
            // synchronous convert-fill into buffer `cur`
#pragma unroll
            for (int stp = 0; stp < 8; ++stp) {
                int f   = stp * 512 + tid * 4;
                int row = f >> 6, col = f & 63;
                const float4 k4 = *(const float4*)(kf + (size_t)(kv0 + row) * DM + h * HD + col);
                uint2 st; st.x = pk2(k4.x, k4.y); st.y = pk2(k4.z, k4.w);
                *(uint2*)&lds_k[cur][f] = st;
                const float4 v4 = *(const float4*)(vf + (size_t)(kv0 + row) * DM + h * HD + col);
                lds_v[cur][(col + 0) * BN + row] = f2bf(v4.x);
                lds_v[cur][(col + 1) * BN + row] = f2bf(v4.y);
                lds_v[cur][(col + 2) * BN + row] = f2bf(v4.z);
                lds_v[cur][(col + 3) * BN + row] = f2bf(v4.w);
            }
        }
        __syncthreads();

        const __bf16* ck = lds_k[cur];
        const __bf16* cv = lds_v[cur];

        // ---- scores: S = Q * K^T ----
        v8f s[4];
#pragma unroll
        for (int t = 0; t < 4; ++t) {
            v8f acc = {};
            const int keyl = t * 16 + ln;
#pragma unroll
            for (int kk = 0; kk < 2; ++kk) {
                v16bf kbv = ld_b16(&ck[keyl * HD + kk * 32 + hs * 8]);
                acc = wmma_bf(qa[kk], kbv, acc);
            }
            s[t] = acc * scale;
        }

        // ---- online softmax ----
        v8f tm = s[0];
#pragma unroll
        for (int t = 1; t < 4; ++t)
#pragma unroll
            for (int r = 0; r < 8; ++r) tm[r] = fmaxf(tm[r], s[t][r]);
#pragma unroll
        for (int m = 1; m < 16; m <<= 1)
#pragma unroll
            for (int r = 0; r < 8; ++r) tm[r] = fmaxf(tm[r], __shfl_xor(tm[r], m, 32));

        float nm[8], corr[8];
#pragma unroll
        for (int r = 0; r < 8; ++r) {
            nm[r]   = fmaxf(mi[r], tm[r]);
            corr[r] = __expf(mi[r] - nm[r]);
            mi[r]   = nm[r];
        }
#pragma unroll
        for (int t = 0; t < 4; ++t)
#pragma unroll
            for (int r = 0; r < 8; ++r) s[t][r] = __expf(s[t][r] - nm[r]);

        v8f rs = s[0] + s[1] + s[2] + s[3];
#pragma unroll
        for (int m = 1; m < 16; m <<= 1)
#pragma unroll
            for (int r = 0; r < 8; ++r) rs[r] += __shfl_xor(rs[r], m, 32);
#pragma unroll
        for (int r = 0; r < 8; ++r) li[r] = li[r] * corr[r] + rs[r];

        // ---- transpose P via per-wave LDS scratch (C-layout -> A-layout) ----
        __bf16* pw = lds_p + wave * (16 * BN);
#pragma unroll
        for (int t = 0; t < 4; ++t)
#pragma unroll
            for (int r = 0; r < 8; ++r)
                pw[(r + 8 * hs) * BN + t * 16 + ln] = f2bf(s[t][r]);

#pragma unroll
        for (int t = 0; t < 4; ++t)
#pragma unroll
            for (int r = 0; r < 8; ++r) O[t][r] *= corr[r];

        v16bf pa[2];
#pragma unroll
        for (int kk = 0; kk < 2; ++kk) {
            const __bf16* pb = pw + ln * BN + kk * 32 + hs * 8;
            v8bf lo = *(const v8bf*)pb;
            v8bf hi = *(const v8bf*)(pb + 16);
            pa[kk] = cat16(lo, hi);
        }

        // ---- O = O*corr + P * V ----
#pragma unroll
        for (int t = 0; t < 4; ++t) {
#pragma unroll
            for (int kk = 0; kk < 2; ++kk) {
                v16bf vb = ld_b16(&cv[(t * 16 + ln) * BN + kk * 32 + hs * 8]);
                O[t] = wmma_bf(pa[kk], vb, O[t]);
            }
        }
        __syncthreads();
    }

    // ---- normalize (reference uses sum + 1e-6) and store ----
    float inv[8];
#pragma unroll
    for (int r = 0; r < 8; ++r) inv[r] = 1.0f / (li[r] + 1e-6f);
#pragma unroll
    for (int t = 0; t < 4; ++t)
#pragma unroll
        for (int r = 0; r < 8; ++r) {
            size_t row = (size_t)(qb * BM + wave * 16 + r + 8 * hs);
            out[row * DM + h * HD + t * 16 + ln] = O[t][r] * inv[r];
        }
}

extern "C" void kernel_launch(void* const* d_in, const int* in_sizes, int n_in,
                              void* d_out, int out_size, void* d_ws, size_t ws_size,
                              hipStream_t stream) {
    (void)n_in; (void)out_size;
    const float* q = (const float*)d_in[0];
    const float* k = (const float*)d_in[1];
    const float* v = (const float*)d_in[2];
    float* out = (float*)d_out;
    const int S = in_sizes[0] / DM;   // 4096
    const size_t elems = (size_t)S * DM;
    const size_t need  = 2 * elems * sizeof(unsigned short);   // kb + vt (16 MB)

    dim3 agrid(S / BM, NH), ablock(128);
    if (ws_size >= need) {
        unsigned short* kb = (unsigned short*)d_ws;
        unsigned short* vt = kb + elems;
        cvt_k_kernel<<<(unsigned)(elems / 1024), 256, 0, stream>>>(k, kb);
        vtrans_kernel<<<dim3(S / 64, DM / 64), 256, 0, stream>>>(v, vt, S);
        fa_fwd_kernel<true><<<agrid, ablock, 0, stream>>>(
            q, nullptr, nullptr, (const __bf16*)kb, (const __bf16*)vt, out, S);
    } else {
        fa_fwd_kernel<false><<<agrid, ablock, 0, stream>>>(
            q, k, v, nullptr, nullptr, out, S);
    }
}
